// CausalSelfAttention_71889162600635
// MI455X (gfx1250) — matrix operand order repaired and checked by reading, not compiled, not run on hardware
//
#include <hip/hip_runtime.h>
#include <hip/hip_bf16.h>

// ---------------------------------------------------------------------------
// Causal self-attention (B=4, T=2048, C=1024, NH=16, HD=64) for gfx1250.
// bf16 WMMA (v_wmma_f32_16x16x32_bf16) for all matmuls, f32 accumulation.
// V tiles staged into LDS with the Tensor Data Mover (tensor_load_to_lds,
// s_wait_tensorcnt), double-buffered across key blocks.
// ---------------------------------------------------------------------------

typedef __bf16  v8bf  __attribute__((ext_vector_type(8)));
typedef __bf16  v16bf __attribute__((ext_vector_type(16)));
typedef float   v8f   __attribute__((ext_vector_type(8)));
typedef unsigned int u32x4 __attribute__((ext_vector_type(4)));
typedef int          i32x4 __attribute__((ext_vector_type(4)));
typedef int          i32x8 __attribute__((ext_vector_type(8)));

constexpr int Bn  = 4;
constexpr int Tn  = 2048;
constexpr int Cn  = 1024;
constexpr int NHn = 16;
constexpr int HDn = 64;
constexpr int Mn  = Bn * Tn;          // 8192 rows of x / y
constexpr int N3C = 3 * Cn;           // 3072

#define DEVINL __device__ __forceinline__

#if defined(__has_builtin)
#  if __has_builtin(__builtin_amdgcn_tensor_load_to_lds) && __has_builtin(__builtin_amdgcn_s_wait_tensorcnt)
#    define HAS_TDM 1
#  else
#    define HAS_TDM 0
#  endif
#else
#  define HAS_TDM 0
#endif

DEVINL v16bf cat16(v8bf lo, v8bf hi) {
    return __builtin_shufflevector(lo, hi, 0,1,2,3,4,5,6,7,8,9,10,11,12,13,14,15);
}

DEVINL v8f wmma_bf16(v16bf a, v16bf b, v8f c) {
    return __builtin_amdgcn_wmma_f32_16x16x32_bf16(false, a, false, b, (short)0, c, false, false);
}

#if HAS_TDM
// TDM: load a 32x64 bf16 tile (rows x HD, row-major, stride HD) from global
// into LDS at byte offset ldsAddr. D# layout per CDNA5 ISA §8.3/§8.4:
//   group0: count=1 | lds_addr | global_addr[56:0] | type=2
//   group1: data_size=2B, tensor_dim0=64, tensor_dim1=32, tile 64x32, stride0=64
DEVINL void tdm_load_v_tile(unsigned ldsAddr, const __bf16* gptr) {
    unsigned long long ga = (unsigned long long)(__SIZE_TYPE__)gptr;
    u32x4 g0 = { 1u,                                   // count=1, user descriptor
                 ldsAddr,                              // LDS byte address
                 (unsigned)(ga & 0xffffffffu),
                 (unsigned)((ga >> 32) & 0x01ffffffu) | 0x80000000u }; // type=2
    i32x8 g1 = { 0x00010000,        // workgroup_mask=0, data_size=1 (2 bytes)
                 (64 << 16),        // tensor_dim0[15:0]=64  (bits 63:48)
                 (32 << 16),        // dim0 hi=0 | tensor_dim1[15:0]=32
                 (64 << 16),        // dim1 hi=0 | tile_dim0=64
                 32,                // tile_dim1=32 | tile_dim2=0
                 64,                // tensor_dim0_stride[31:0]=64
                 0,                 // stride0 hi | tensor_dim1_stride lo
                 0 };
    i32x4 z4 = { 0, 0, 0, 0 };
#if __clang_major__ >= 23
    i32x8 z8 = { 0, 0, 0, 0, 0, 0, 0, 0 };
    __builtin_amdgcn_tensor_load_to_lds(g0, g1, z4, z4, z8, 0);
#else
    __builtin_amdgcn_tensor_load_to_lds(g0, g1, z4, z4, 0);
#endif
}
#endif

// --------------------------- cast / transpose ------------------------------

__global__ __launch_bounds__(256) void cast_bf16_kernel(const float* __restrict__ in,
                                                        __bf16* __restrict__ out, int n) {
    int i = blockIdx.x * 256 + threadIdx.x;
    if (i < n) out[i] = (__bf16)in[i];
}

// in: f32 [K][N] row-major  ->  out: bf16 [N][K] (i.e. B^T, K contiguous)
__global__ __launch_bounds__(256) void transpose_bf16_kernel(const float* __restrict__ in,
                                                             __bf16* __restrict__ out,
                                                             int K, int N) {
    int i = blockIdx.x * 256 + threadIdx.x;
    if (i >= K * N) return;
    int k = i / N;
    int n = i - k * N;
    out[(size_t)n * K + k] = (__bf16)in[i];
}

// ------------------------------- QKV GEMM ----------------------------------
// A: bf16 [Mn][Cn]; BT: bf16 [N3C][Cn]; bias f32 [N3C].
// Wave computes a 32x32 tile (2 A-frags x 2 B-frags -> 4 accums, 2x register
// reuse of each fragment); block = 8 waves = 64x128 tile.
// Output scattered to q/k/v as [B][NH][T][HD] bf16.

__global__ __launch_bounds__(256) void gemm_qkv_kernel(const __bf16* __restrict__ A,
                                                       const __bf16* __restrict__ BT,
                                                       const float* __restrict__ bias,
                                                       __bf16* __restrict__ qb,
                                                       __bf16* __restrict__ kb,
                                                       __bf16* __restrict__ vb) {
    const int lane = threadIdx.x & 31;
    const int w    = threadIdx.x >> 5;
    const int sel  = lane >> 4;
    const int lr   = lane & 15;
    const int m0   = blockIdx.x * 64  + (w >> 2) * 32;
    const int n0   = blockIdx.y * 128 + (w & 3) * 32;

    const __bf16* a0p = A  + (size_t)(m0 + lr) * Cn + sel * 8;
    const __bf16* a1p = a0p + (size_t)16 * Cn;
    const __bf16* b0p = BT + (size_t)(n0 + lr) * Cn + sel * 8;
    const __bf16* b1p = b0p + (size_t)16 * Cn;

    v8f acc00 = {}, acc01 = {}, acc10 = {}, acc11 = {};
    for (int kk = 0; kk < Cn; kk += 32) {
        v16bf af0 = cat16(*(const v8bf*)(a0p + kk), *(const v8bf*)(a0p + kk + 16));
        v16bf af1 = cat16(*(const v8bf*)(a1p + kk), *(const v8bf*)(a1p + kk + 16));
        v16bf bf0 = cat16(*(const v8bf*)(b0p + kk), *(const v8bf*)(b0p + kk + 16));
        v16bf bf1 = cat16(*(const v8bf*)(b1p + kk), *(const v8bf*)(b1p + kk + 16));
        acc00 = wmma_bf16(af0, bf0, acc00);
        acc01 = wmma_bf16(af0, bf1, acc01);
        acc10 = wmma_bf16(af1, bf0, acc10);
        acc11 = wmma_bf16(af1, bf1, acc11);
    }

#pragma unroll
    for (int nt = 0; nt < 2; ++nt) {
        const int ncol  = n0 + nt * 16 + lr;
        const int which = ncol >> 10;
        const int cc    = ncol & (Cn - 1);
        const int h     = cc >> 6;
        const int hd    = cc & (HDn - 1);
        __bf16* dst = (which == 0) ? qb : ((which == 1) ? kb : vb);
        const float bv = bias[ncol];
#pragma unroll
        for (int mt = 0; mt < 2; ++mt) {
            const v8f& acc = (mt == 0) ? (nt == 0 ? acc00 : acc01)
                                       : (nt == 0 ? acc10 : acc11);
#pragma unroll
            for (int r = 0; r < 8; ++r) {
                int m  = m0 + mt * 16 + r + sel * 8;
                int bb = m >> 11;               // / Tn
                int t  = m & (Tn - 1);
                dst[((size_t)(bb * NHn + h) * Tn + t) * HDn + hd] = (__bf16)(acc[r] + bv);
            }
        }
    }
}

// --------------------------- flash attention -------------------------------
// One wave owns one (b,h,q-block of 16 rows); block = 4 waves (128 threads).
// Streams 32-key blocks with online softmax. V tiles are staged into LDS by
// the Tensor Data Mover, double-buffered (prefetch next block while current
// one is consumed); P relayout goes through per-wave LDS.

__global__ __launch_bounds__(128) void attn_kernel(const __bf16* __restrict__ qb,
                                                   const __bf16* __restrict__ kb,
                                                   const __bf16* __restrict__ vb,
                                                   __bf16* __restrict__ yb) {
    __shared__ __align__(16) __bf16 lds_p[4][16 * 32];       // per-wave P tile
    __shared__ __align__(128) __bf16 lds_vraw[4][2][32 * 64]; // per-wave V, 2 buffers

    const int lane = threadIdx.x & 31;
    const int w    = threadIdx.x >> 5;
    const int sel  = lane >> 4;
    const int lr   = lane & 15;
    const int gid  = blockIdx.x * 4 + w;
    const int iq   = gid & 127;         // Tn/16 q-blocks per (b,h)
    const int bh   = gid >> 7;
    const int q0   = iq * 16;

    const __bf16* Q  = qb + (size_t)bh * Tn * HDn;
    const __bf16* Kp = kb + (size_t)bh * Tn * HDn;
    const __bf16* Vp = vb + (size_t)bh * Tn * HDn;
    __bf16* Plds = lds_p[w];
    __bf16* vbuf0 = &lds_vraw[w][0][0];
    __bf16* vbuf1 = &lds_vraw[w][1][0];

    // Q A-fragments, resident for the whole loop (HD=64 -> 2 K-steps of 32).
    v16bf aq0, aq1;
    {
        const __bf16* qr = Q + (size_t)(q0 + lr) * HDn;
        aq0 = cat16(*(const v8bf*)(qr + sel * 16),      *(const v8bf*)(qr + 8 + sel * 16));
        aq1 = cat16(*(const v8bf*)(qr + 32 + sel * 16), *(const v8bf*)(qr + 40 + sel * 16));
    }

    v8f o0 = {}, o1 = {}, o2 = {}, o3 = {};
    float mrow[8], lrow[8];
#pragma unroll
    for (int r = 0; r < 8; ++r) { mrow[r] = -__builtin_inff(); lrow[r] = 0.0f; }

    const float scale = 0.125f;                       // 1/sqrt(64)
    const int nkb = (q0 + 16 + 31) >> 5;              // causal key-block count

#if HAS_TDM
    // Prime the pipeline: TDM-stage V block 0 into buffer 0.
    tdm_load_v_tile((unsigned)(__SIZE_TYPE__)vbuf0, Vp);
#endif

    for (int kbk = 0; kbk < nkb; ++kbk) {
        const int k0 = kbk * 32;

        // ---- S = Q @ K^T for 32 keys (two 16-col halves, two K-steps each)
        const __bf16* kr0 = Kp + (size_t)(k0 + lr) * HDn + sel * 16;
        const __bf16* kr1 = Kp + (size_t)(k0 + 16 + lr) * HDn + sel * 16;
        v16bf bk00 = *(const v16bf*)(kr0);
        v16bf bk01 = *(const v16bf*)(kr0 + 32);
        v16bf bk10 = *(const v16bf*)(kr1);
        v16bf bk11 = *(const v16bf*)(kr1 + 32);

        v8f s0 = {}, s1 = {};
        s0 = wmma_bf16(aq0, bk00, s0);
        s0 = wmma_bf16(aq1, bk01, s0);
        s1 = wmma_bf16(aq0, bk10, s1);
        s1 = wmma_bf16(aq1, bk11, s1);

        // ---- V staging: prefetch next block, wait for current
        const __bf16* raw;
#if HAS_TDM
        if (kbk + 1 < nkb) {
            tdm_load_v_tile((unsigned)(__SIZE_TYPE__)(((kbk + 1) & 1) ? vbuf1 : vbuf0),
                            Vp + (size_t)(k0 + 32) * HDn);
            __builtin_amdgcn_s_wait_tensorcnt(1);   // current tile done (in-order)
        } else {
            __builtin_amdgcn_s_wait_tensorcnt(0);
        }
        raw = (kbk & 1) ? vbuf1 : vbuf0;
#else
        {   // fallback: cooperative copy, lane owns key row (k0+lane)
            const __bf16* vr = Vp + (size_t)(k0 + lane) * HDn;
            __bf16* dst = vbuf0 + lane * HDn;
#pragma unroll
            for (int j = 0; j < 8; ++j)
                *(v8bf*)(dst + j * 8) = *(const v8bf*)(vr + j * 8);
        }
        raw = vbuf0;
#endif
        __builtin_amdgcn_wave_barrier();

        // ---- online softmax over the 16x32 score tile
        float alpha_r[8];
#pragma unroll
        for (int r = 0; r < 8; ++r) {
            const int qrow = q0 + r + sel * 8;
            const int key0 = k0 + lr;
            float x0 = s0[r] * scale; if (key0 > qrow)      x0 = -__builtin_inff();
            float x1 = s1[r] * scale; if (key0 + 16 > qrow) x1 = -__builtin_inff();
            float mx = fmaxf(x0, x1);
            mx = fmaxf(mx, __shfl_xor(mx, 1, 16));
            mx = fmaxf(mx, __shfl_xor(mx, 2, 16));
            mx = fmaxf(mx, __shfl_xor(mx, 4, 16));
            mx = fmaxf(mx, __shfl_xor(mx, 8, 16));
            const float mold = mrow[r];
            const float mnew = fmaxf(mold, mx);
            float a, p0, p1;
            if (mnew == -__builtin_inff()) { a = 1.0f; p0 = 0.0f; p1 = 0.0f; }
            else {
                a  = __expf(mold - mnew);   // exp(-inf)=0 covers the first update
                p0 = __expf(x0 - mnew);
                p1 = __expf(x1 - mnew);
            }
            float rs = p0 + p1;
            rs += __shfl_xor(rs, 1, 16);
            rs += __shfl_xor(rs, 2, 16);
            rs += __shfl_xor(rs, 4, 16);
            rs += __shfl_xor(rs, 8, 16);
            lrow[r]    = lrow[r] * a + rs;
            mrow[r]    = mnew;
            alpha_r[r] = a;
            // D-layout (lane = key col, vgpr = q row) -> row-major P in LDS
            Plds[(r + sel * 8) * 32 + lr]      = (__bf16)p0;
            Plds[(r + sel * 8) * 32 + lr + 16] = (__bf16)p1;
        }

#pragma unroll
        for (int r = 0; r < 8; ++r) {
            o0[r] *= alpha_r[r]; o1[r] *= alpha_r[r];
            o2[r] *= alpha_r[r]; o3[r] *= alpha_r[r];
        }

        __builtin_amdgcn_wave_barrier();   // LDS stores before the reads

        // ---- O += P @ V  (K = 32 keys, 4 hd chunks of 16)
        v16bf pf;
        {
            const __bf16* pr = Plds + lr * 32 + sel * 16;
            pf = cat16(*(const v8bf*)(pr), *(const v8bf*)(pr + 8));
        }
        // B-fragments gathered from the row-major staged tile (V^T access).
#pragma unroll
        for (int c = 0; c < 4; ++c) {
            v16bf vf;
#pragma unroll
            for (int j = 0; j < 16; ++j)
                vf[j] = raw[(sel * 16 + j) * HDn + c * 16 + lr];
            if (c == 0) o0 = wmma_bf16(pf, vf, o0);
            if (c == 1) o1 = wmma_bf16(pf, vf, o1);
            if (c == 2) o2 = wmma_bf16(pf, vf, o2);
            if (c == 3) o3 = wmma_bf16(pf, vf, o3);
        }
        __builtin_amdgcn_wave_barrier();
    }

    // ---- epilogue: normalize and write y back as [B][T][C] bf16
    const int bb = bh >> 4;      // / NHn
    const int h  = bh & 15;
#pragma unroll
    for (int r = 0; r < 8; ++r) {
        const float inv = 1.0f / lrow[r];
        const int t = q0 + r + sel * 8;
        const size_t base = ((size_t)(bb * Tn + t)) * Cn + h * HDn + lr;
        yb[base + 0]  = (__bf16)(o0[r] * inv);
        yb[base + 16] = (__bf16)(o1[r] * inv);
        yb[base + 32] = (__bf16)(o2[r] * inv);
        yb[base + 48] = (__bf16)(o3[r] * inv);
    }
}

// ------------------------------ proj GEMM ----------------------------------
// A: bf16 [Mn][Cn]; BT: bf16 [Cn][Cn]; bias f32 [Cn]; out f32 [Mn][Cn].
// Same 32x32-per-wave tiling as the QKV GEMM.

__global__ __launch_bounds__(256) void gemm_proj_kernel(const __bf16* __restrict__ A,
                                                        const __bf16* __restrict__ BT,
                                                        const float* __restrict__ bias,
                                                        float* __restrict__ out) {
    const int lane = threadIdx.x & 31;
    const int w    = threadIdx.x >> 5;
    const int sel  = lane >> 4;
    const int lr   = lane & 15;
    const int m0   = blockIdx.x * 64  + (w >> 2) * 32;
    const int n0   = blockIdx.y * 128 + (w & 3) * 32;

    const __bf16* a0p = A  + (size_t)(m0 + lr) * Cn + sel * 8;
    const __bf16* a1p = a0p + (size_t)16 * Cn;
    const __bf16* b0p = BT + (size_t)(n0 + lr) * Cn + sel * 8;
    const __bf16* b1p = b0p + (size_t)16 * Cn;

    v8f acc00 = {}, acc01 = {}, acc10 = {}, acc11 = {};
    for (int kk = 0; kk < Cn; kk += 32) {
        v16bf af0 = cat16(*(const v8bf*)(a0p + kk), *(const v8bf*)(a0p + kk + 16));
        v16bf af1 = cat16(*(const v8bf*)(a1p + kk), *(const v8bf*)(a1p + kk + 16));
        v16bf bf0 = cat16(*(const v8bf*)(b0p + kk), *(const v8bf*)(b0p + kk + 16));
        v16bf bf1 = cat16(*(const v8bf*)(b1p + kk), *(const v8bf*)(b1p + kk + 16));
        acc00 = wmma_bf16(af0, bf0, acc00);
        acc01 = wmma_bf16(af0, bf1, acc01);
        acc10 = wmma_bf16(af1, bf0, acc10);
        acc11 = wmma_bf16(af1, bf1, acc11);
    }

#pragma unroll
    for (int nt = 0; nt < 2; ++nt) {
        const int n = n0 + nt * 16 + lr;
        const float bv = bias[n];
#pragma unroll
        for (int mt = 0; mt < 2; ++mt) {
            const v8f& acc = (mt == 0) ? (nt == 0 ? acc00 : acc01)
                                       : (nt == 0 ? acc10 : acc11);
#pragma unroll
            for (int r = 0; r < 8; ++r) {
                int m = m0 + mt * 16 + r + sel * 8;
                out[(size_t)m * Cn + n] = acc[r] + bv;
            }
        }
    }
}

// ------------------------------- launcher ----------------------------------

extern "C" void kernel_launch(void* const* d_in, const int* in_sizes, int n_in,
                              void* d_out, int out_size, void* d_ws, size_t ws_size,
                              hipStream_t stream) {
    const float* x      = (const float*)d_in[0];
    const float* w_attn = (const float*)d_in[1];
    const float* b_attn = (const float*)d_in[2];
    const float* w_proj = (const float*)d_in[3];
    const float* b_proj = (const float*)d_in[4];
    float* out = (float*)d_out;

    char* ws = (char*)d_ws;
    const size_t elems_xc = (size_t)Mn * Cn;          // 8388608
    __bf16* xbf = (__bf16*)ws;                 ws += elems_xc * 2;
    __bf16* waT = (__bf16*)ws;                 ws += (size_t)N3C * Cn * 2;
    __bf16* wpT = (__bf16*)ws;                 ws += (size_t)Cn * Cn * 2;
    __bf16* qb  = (__bf16*)ws;                 ws += elems_xc * 2;
    __bf16* kb  = (__bf16*)ws;                 ws += elems_xc * 2;
    __bf16* vb  = (__bf16*)ws;                 ws += elems_xc * 2;
    __bf16* yb  = (__bf16*)ws;                 ws += elems_xc * 2;

    // 1. cast / transpose to bf16
    cast_bf16_kernel<<<(int)(elems_xc / 256), 256, 0, stream>>>(x, xbf, (int)elems_xc);
    transpose_bf16_kernel<<<(Cn * N3C) / 256, 256, 0, stream>>>(w_attn, waT, Cn, N3C);
    transpose_bf16_kernel<<<(Cn * Cn) / 256, 256, 0, stream>>>(w_proj, wpT, Cn, Cn);

    // 2. QKV projection -> per-head bf16 Q/K/V
    gemm_qkv_kernel<<<dim3(Mn / 64, N3C / 128), 256, 0, stream>>>(xbf, waT, b_attn, qb, kb, vb);

    // 3. causal flash attention (TDM-staged V) -> y (bf16 [B,T,C])
    attn_kernel<<<(Bn * NHn * (Tn / 16)) / 4, 128, 0, stream>>>(qb, kb, vb, yb);

    // 4. output projection -> f32 out
    gemm_proj_kernel<<<dim3(Mn / 64, Cn / 128), 256, 0, stream>>>(yb, wpT, b_proj, out);
}
